// QLinear_979252543653
// MI455X (gfx1250) — 8-bit matrix operand order repaired and checked by reading, compiled, not run on hardware
//
#include <hip/hip_runtime.h>

#define NROW 8192
#define DIN  2048
#define DOUT 2048

typedef __attribute__((ext_vector_type(8))) int v8i;
typedef __attribute__((ext_vector_type(4))) int v4i;

__global__ __launch_bounds__(256) void quantize_kernel(
    const float* __restrict__ x, const float* __restrict__ scale_p,
    const float* __restrict__ zp_p, unsigned char* __restrict__ xq)
{
    unsigned i = blockIdx.x * 256u + threadIdx.x;
    const float s  = scale_p[0];
    const float zp = zp_p[0];
    const float inv = 1.0f / s;
    float4 v = ((const float4*)x)[i];
    float f0 = fminf(fmaxf(rintf(v.x * inv) + zp, 0.0f), 255.0f);
    float f1 = fminf(fmaxf(rintf(v.y * inv) + zp, 0.0f), 255.0f);
    float f2 = fminf(fmaxf(rintf(v.z * inv) + zp, 0.0f), 255.0f);
    float f3 = fminf(fmaxf(rintf(v.w * inv) + zp, 0.0f), 255.0f);
    unsigned p = (unsigned)f0 | ((unsigned)f1 << 8) |
                 ((unsigned)f2 << 16) | ((unsigned)f3 << 24);
    ((unsigned*)xq)[i] = p;
}

__global__ __launch_bounds__(256) void transpose_kernel(
    const unsigned char* __restrict__ qk, unsigned char* __restrict__ ktr)
{
    __shared__ unsigned char t[64][68];
    const int k0 = blockIdx.x * 64;
    const int n0 = blockIdx.y * 64;
    const int tid = threadIdx.x;
    const int r = tid >> 2;
    const int c = (tid & 3) * 16;

    union { uint4 v; unsigned char b[16]; } u;
    u.v = *(const uint4*)(qk + (size_t)(k0 + r) * DOUT + n0 + c);
#pragma unroll
    for (int i = 0; i < 16; ++i) t[r][c + i] = u.b[i];
    __syncthreads();
    union { uint4 v; unsigned char b[16]; } w;
#pragma unroll
    for (int i = 0; i < 16; ++i) w.b[i] = t[c + i][r];
    *(uint4*)(ktr + (size_t)(n0 + r) * DIN + k0 + c) = w.v;
}

#define LDS_STRIDE 80
#define TILE_BYTES (128 * LDS_STRIDE)
#define BUF_BYTES  (2 * TILE_BYTES)

#if __has_builtin(__builtin_amdgcn_tensor_load_to_lds) && \
    __has_builtin(__builtin_amdgcn_s_wait_tensorcnt)
#define HAVE_TDM 1
typedef __attribute__((ext_vector_type(4))) unsigned int tdm_v4u;
typedef __attribute__((ext_vector_type(8))) int          tdm_v8i;
typedef __attribute__((ext_vector_type(4))) int          tdm_v4i;

__device__ inline void tdm_load_tile(unsigned long long ga, unsigned lds_off,
                                     unsigned tensor_rows)
{
    tdm_v4u g0 = { 1u,
                   lds_off,
                   (unsigned)(ga & 0xFFFFFFFFull),
                   (unsigned)((ga >> 32) & 0x01FFFFFFull) | (2u << 30) };
    tdm_v8i g1 = { (int)((1u << 20) | (3u << 22) | (3u << 25)),
                   (int)((DIN & 0xFFFFu) << 16),
                   (int)(((tensor_rows & 0xFFFFu) << 16) | ((unsigned)DIN >> 16)),
                   (int)((64u << 16) | (tensor_rows >> 16)),
                   (int)128u,
                   (int)DIN,
                   0, 0 };
    tdm_v4i gz = { 0, 0, 0, 0 };
#if __clang_major__ >= 23
    tdm_v8i gz8 = { 0, 0, 0, 0, 0, 0, 0, 0 };
    __builtin_amdgcn_tensor_load_to_lds(g0, g1, gz, gz, gz8, 0);
#else
    __builtin_amdgcn_tensor_load_to_lds(g0, g1, gz, gz, 0);
#endif
}
#endif

__global__ __launch_bounds__(256) void qgemm_kernel(
    const unsigned char* __restrict__ xq,
    const unsigned char* __restrict__ ktr,
    const unsigned short* __restrict__ qbias,
    const float* __restrict__ scale_p, const float* __restrict__ zp_p,
    float* __restrict__ out)
{
    __shared__ __attribute__((aligned(16))) unsigned char smem[2 * BUF_BYTES];

    const int tid  = threadIdx.x;
    const int lane = tid & 31;
    const int wave = tid >> 5;
    const int wm = wave & 3;
    const int wn = wave >> 2;

    const int m0 = blockIdx.y * 128;
    const int n0 = blockIdx.x * 128;

    const unsigned char* gA = xq  + (size_t)m0 * DIN;
    const unsigned char* gB = ktr + (size_t)n0 * DIN;

#ifdef HAVE_TDM
    auto issue = [&](unsigned lb, int k0) {
        if (wave == 0) {
            tdm_load_tile((unsigned long long)(gA + k0),
                          (unsigned)(unsigned long long)&smem[lb], NROW);
            tdm_load_tile((unsigned long long)(gB + k0),
                          (unsigned)(unsigned long long)&smem[lb + TILE_BYTES],
                          DOUT);
        }
    };
#else
    auto issue = [&](unsigned lb, int k0) {
#pragma unroll
        for (int i = 0; i < 2; ++i) {
            int t = tid + i * 256;
            int r = t >> 2;
            int c = (t & 3) << 4;
            unsigned la =
                (unsigned)(unsigned long long)&smem[lb + r * LDS_STRIDE + c];
            unsigned long long ga =
                (unsigned long long)(gA + (size_t)r * DIN + k0 + c);
            asm volatile("global_load_async_to_lds_b128 %0, %1, off"
                         :: "v"(la), "v"(ga) : "memory");
            unsigned lbd = (unsigned)(unsigned long long)
                &smem[lb + TILE_BYTES + r * LDS_STRIDE + c];
            unsigned long long gb =
                (unsigned long long)(gB + (size_t)r * DIN + k0 + c);
            asm volatile("global_load_async_to_lds_b128 %0, %1, off"
                         :: "v"(lbd), "v"(gb) : "memory");
        }
    };
#endif

    v8i acc[2][4] = {};

    issue(0, 0);
    unsigned cur = 0;

    const int lrow = lane & 15;
    const int ka8  = (lane >> 4) * 16;
    const int kb16 = (lane >> 4) * 16;

    for (int kt = 0; kt < DIN / 64; ++kt) {
#ifdef HAVE_TDM
        if (wave == 0) __builtin_amdgcn_s_wait_tensorcnt(0);
#else
        asm volatile("s_wait_asynccnt 0x0" ::: "memory");
#endif
        __syncthreads();
        if (kt + 1 < DIN / 64) issue(cur ^ (unsigned)BUF_BYTES, (kt + 1) * 64);

        const unsigned char* sA = &smem[cur];
        const unsigned char* sB = &smem[cur + TILE_BYTES];

        v8i af[2];
#pragma unroll
        for (int sm = 0; sm < 2; ++sm) {
            const unsigned char* p =
                sA + (wm * 32 + sm * 16 + lrow) * LDS_STRIDE + ka8;
            union { v8i v; unsigned long long q[4]; } u;
            u.q[0] = *(const unsigned long long*)(p);
            u.q[1] = *(const unsigned long long*)(p + 8);
            u.q[2] = *(const unsigned long long*)(p + 32);
            u.q[3] = *(const unsigned long long*)(p + 40);
            af[sm] = u.v;
        }
        v8i bf[4];
#pragma unroll
        for (int sn = 0; sn < 4; ++sn) {
            const unsigned char* p =
                sB + (wn * 64 + sn * 16 + lrow) * LDS_STRIDE + kb16;
            v4i lo = *(const v4i*)(p);
            v4i hi = *(const v4i*)(p + 32);
            bf[sn] = __builtin_shufflevector(lo, hi, 0, 1, 2, 3, 4, 5, 6, 7);
        }
#pragma unroll
        for (int sm = 0; sm < 2; ++sm)
#pragma unroll
            for (int sn = 0; sn < 4; ++sn)
                acc[sm][sn] = __builtin_amdgcn_wmma_i32_16x16x64_iu8(
                    false, af[sm], false, bf[sn], acc[sm][sn], false, false);

        cur ^= (unsigned)BUF_BYTES;
    }

    const float sc = scale_p[0];
    const float zp = zp_p[0];
    const int colbase = n0 + wn * 64;
    const int rowlane = (lane >> 4) * 8;
#pragma unroll
    for (int sm = 0; sm < 2; ++sm) {
#pragma unroll
        for (int sn = 0; sn < 4; ++sn) {
            union { v8i v; int s[8]; } u;
            u.v = acc[sm][sn];
            const int col = colbase + sn * 16 + (lane & 15);
            const unsigned bias = qbias[col];
            const int rb = m0 + wm * 32 + sm * 16 + rowlane;
#pragma unroll
            for (int j = 0; j < 8; ++j) {
                unsigned t = ((unsigned)u.s[j] + bias) & 0xFFFFu;
                out[(size_t)(rb + j) * DOUT + col] = ((float)t - zp) * sc;
            }
        }
    }
}

extern "C" void kernel_launch(void* const* d_in, const int* in_sizes, int n_in,
                              void* d_out, int out_size, void* d_ws, size_t ws_size,
                              hipStream_t stream) {
    (void)in_sizes; (void)n_in; (void)out_size; (void)ws_size;
    const float*          x     = (const float*)d_in[0];
    const float*          scale = (const float*)d_in[1];
    const float*          zp    = (const float*)d_in[2];
    const unsigned char*  qk    = (const unsigned char*)d_in[3];
    const unsigned short* qb    = (const unsigned short*)d_in[4];
    float* out = (float*)d_out;

    unsigned char* xq  = (unsigned char*)d_ws;
    unsigned char* ktr = xq + (size_t)NROW * DIN;

    quantize_kernel<<<(NROW * DIN / 4) / 256, 256, 0, stream>>>(x, scale, zp, xq);
    transpose_kernel<<<dim3(DIN / 64, DOUT / 64), 256, 0, stream>>>(qk, ktr);
    qgemm_kernel<<<dim3(DOUT / 128, NROW / 128), 256, 0, stream>>>(
        xq, ktr, qb, scale, zp, out);
}
